// MaskEmbdMultiMPN_NNConv_v3_39728447488424
// MI455X (gfx1250) — matrix operand order repaired and checked by reading, not compiled, not run on hardware
//
#include <hip/hip_runtime.h>
#include <stdint.h>

// ---------------- problem constants ----------------
#define NN    20000
#define EE    50000
#define E2    100000      // 2E undirected duplication
#define HH    32
#define FNN   4
#define FEE   2
#define FOO   4
#define LL    3
#define TILES (E2 / 16)   // 6250 edge tiles of 16 edges

// Bsw main region: [L][32 kk][2 nt][32 lane][8 v] dwords
#define BSW_MAIN_DW  (LL * 32 * 2 * 32 * 8)   // 49152 (16384 dw = 64KB per layer)
#define BSW_LAYER_DW (32 * 2 * 32 * 8)        // 16384
// Bias region: [L][2 nt][32 lane][8 v]
#define BSW_BIAS_DW  (LL * 2 * 32 * 8)        // 1536

typedef __attribute__((ext_vector_type(16))) __bf16 v16bf;
typedef __attribute__((ext_vector_type(8)))  float  v8f;
typedef uint32_t v4u __attribute__((ext_vector_type(4)));
typedef int      v8i_t __attribute__((ext_vector_type(8)));
typedef int      v4i_t __attribute__((ext_vector_type(4)));

__device__ __forceinline__ unsigned short f2bf(float f) {
    unsigned int u = __float_as_uint(f);
    unsigned int r = u + 0x7FFFu + ((u >> 16) & 1u);   // round-to-nearest-even
    return (unsigned short)(r >> 16);
}

__device__ __forceinline__ float gelu_exact(float x) {
    return 0.5f * x * (1.0f + erff(x * 0.70710678118654752440f));
}

// packed bf16 * packed bf16 (VOP3P v_pk_mul_bf16)
__device__ __forceinline__ uint32_t pk_mul_bf16(uint32_t a, uint32_t b) {
    uint32_t d;
    asm("v_pk_mul_bf16 %0, %1, %2" : "=v"(d) : "v"(a), "v"(b));
    return d;
}

// pack two f32 -> packed bf16 dword (lo in [15:0], hi in [31:16])
__device__ __forceinline__ uint32_t cvt_pk_bf16(float lo, float hi) {
    uint32_t d;
    asm("v_cvt_pk_bf16_f32 %0, %1, %2" : "=v"(d) : "v"(lo), "v"(hi));
    return d;
}

// ---------------- B-operand swizzle prep (bf16, ISA 7.12.2 B layout) ----
// B chunk is 32x16 (KxN): lane n<16 holds N=n, K {0..7,16..23}; lanes>=16 hold K {8..15,24..31}.
__global__ void prep_b_kernel(const float* __restrict__ eW2,
                              const float* __restrict__ eb2,
                              uint32_t* __restrict__ Bsw,
                              uint32_t* __restrict__ BswBias) {
    int idx = blockIdx.x * blockDim.x + threadIdx.x;
    if (idx < BSW_MAIN_DW) {
        int v    = idx & 7;
        int lane = (idx >> 3) & 31;
        int nt   = (idx >> 8) & 1;
        int kk   = (idx >> 9) & 31;
        int l    = idx >> 14;
        int k0   = ((lane < 16) ? 0 : 8) + 2 * v + ((v >= 4) ? 8 : 0);
        int o    = nt * 16 + (lane & 15);
        const float* w2 = eW2 + (size_t)(l * 32 + kk) * 1024;
        unsigned short lo = f2bf(w2[k0 * 32 + o]);
        unsigned short hi = f2bf(w2[(k0 + 1) * 32 + o]);
        Bsw[idx] = (uint32_t)lo | ((uint32_t)hi << 16);
    } else if (idx < BSW_MAIN_DW + BSW_BIAS_DW) {
        int j    = idx - BSW_MAIN_DW;
        int v    = j & 7;
        int lane = (j >> 3) & 31;
        int nt   = (j >> 8) & 1;
        int l    = j >> 9;
        int k0   = ((lane < 16) ? 0 : 8) + 2 * v + ((v >= 4) ? 8 : 0);
        int o    = nt * 16 + (lane & 15);
        const float* b2 = eb2 + (size_t)l * 1024;
        unsigned short lo = f2bf(b2[k0 * 32 + o]);
        unsigned short hi = f2bf(b2[(k0 + 1) * 32 + o]);
        BswBias[j] = (uint32_t)lo | ((uint32_t)hi << 16);
    }
}

// ---------------- zero scratch ----------------
__global__ void zero_kernel(float* __restrict__ p, int n) {
    int i = blockIdx.x * blockDim.x + threadIdx.x;
    if (i < n) p[i] = 0.0f;
}

// ---------------- node input encoder: h = x@W_in + b_in + MLP(mask) ----
__global__ void node_init_kernel(const float* __restrict__ x,
                                 const float* __restrict__ pm,
                                 const float* __restrict__ W_in,
                                 const float* __restrict__ b_in,
                                 const float* __restrict__ mW1,
                                 const float* __restrict__ mb1,
                                 const float* __restrict__ mW2,
                                 const float* __restrict__ mb2,
                                 float* __restrict__ h) {
    int gid = blockIdx.x * blockDim.x + threadIdx.x;
    int n = gid >> 5, o = gid & 31;
    if (n >= NN) return;
    const float* xr = x + n * FNN;
    const float* pr = pm + n * FNN;
    float z = mb1[o];
#pragma unroll
    for (int f = 0; f < FNN; f++) z += pr[f] * mW1[f * HH + o];
    float m = gelu_exact(z);
    float acc = b_in[o] + mb2[o];
#pragma unroll
    for (int f = 0; f < FNN; f++) acc += xr[f] * W_in[f * HH + o];
#pragma unroll
    for (int j = 0; j < HH; j++) acc += __shfl(m, j, 32) * mW2[j * HH + o];
    h[(size_t)n * HH + o] = acc;
}

// ---------------- edge kernel: fused t=gelu(ea@w1+b1), msg=(t (x) h_src)@w2, scatter ----
__global__ __launch_bounds__(256) void edge_kernel(
        const float* __restrict__ h, float* __restrict__ agg,
        const int* __restrict__ EI,            // [2,E] flat int32
        const float* __restrict__ edge_attr,   // [E,2]
        const float* __restrict__ eW1,         // [L,2,32]
        const float* __restrict__ eb1,         // [L,32]
        const uint32_t* __restrict__ Bsw,
        const uint32_t* __restrict__ BswBias,
        int layer) {
    __shared__ uint32_t sB[BSW_LAYER_DW];   // 64 KB: this layer's swizzled w2 (bf16)

    int wave = threadIdx.x >> 5;
    int lane = threadIdx.x & 31;
    int tile = blockIdx.x * 8 + wave;
    bool active = (tile < TILES);

    // ---- stage this layer's B into LDS with the Tensor Data Mover (wave 0) ----
    // D# per ISA 8.3/8.4: count=1, type=2 (image), data_size=2 (4B elements),
    // 1-row tile of 16384 elements -> linear 64KB global->LDS copy; lds_addr=0
    // (sB is the only LDS allocation). TENSORcnt waited before the barrier.
    if (wave == 0) {
        uint64_t ga = (uint64_t)(uintptr_t)(Bsw + (size_t)layer * BSW_LAYER_DW);
        v4u g0 = { 1u,                                   // count=1 (valid user D#)
                   0u,                                   // lds_addr = 0
                   (uint32_t)ga,                         // global_addr[31:0]
                   (uint32_t)((ga >> 32) & 0x01FFFFFFu) | 0x80000000u }; // addr[56:32] | type=2
        v8i_t g1 = { (int)0x00020000,                    // data_size=2 (4B), no mask/iter/pad
                     (int)0x40000000,                    // tensor_dim0[15:0]=16384 << 16
                     (int)0x00010000,                    // tensor_dim1=1
                     (int)0x40000000,                    // tile_dim0=16384 << 16
                     1,                                  // tile_dim1=1, tile_dim2=0
                     16384,                              // tensor_dim0_stride[31:0]
                     0, 0 };
        v4i_t z4 = { 0, 0, 0, 0 };
        v8i_t z8 = { 0, 0, 0, 0, 0, 0, 0, 0 };
        __builtin_amdgcn_tensor_load_to_lds(g0, g1, z4, z4, z8, 0);
    }

    // ---- per-edge setup (independent of LDS; overlaps the TDM transfer) ----
    int M    = lane & 15;
    int half = lane >> 4;            // 0: k 0..15, 1: k 16..31
    uint32_t tspk[32];
    uint32_t hspk[8];
    int src = 0;
    if (active) {
        int e = tile * 16 + M;
        src = EI[e];                                    // src = EIflat[e] for all e in [0,2E)
        int ea_row = (e < EE) ? e : (e - EE);
        float ea0 = edge_attr[ea_row * 2 + 0];
        float ea1 = edge_attr[ea_row * 2 + 1];

        // t = gelu(ea @ w1 + b1): the 32 k's split across the two half-waves
        // (each half computes 16 erf's for its row), stored as duplicated-bf16 dwords.
        const float* w1p = eW1 + layer * (FEE * HH);
        const float* b1p = eb1 + layer * HH;
        uint32_t td[16];
#pragma unroll
        for (int j = 0; j < 16; j++) {
            int k = half * 16 + j;
            float z = ea0 * w1p[k] + ea1 * w1p[HH + k] + b1p[k];
            float g = gelu_exact(z);
            td[j] = cvt_pk_bf16(g, g);   // bf16 value duplicated in both halves
        }
        // exchange: lane needs t[row=M][kk]; kk<16 lives in lane M, kk>=16 in lane M+16
        int srcLo = M, srcHi = M + 16;
#pragma unroll
        for (int j = 0; j < 32; j++) {
            tspk[j] = (uint32_t)__shfl((int)td[j & 15], (j < 16) ? srcLo : srcHi, 32);
        }

        // gather h[src] in the 16-bit A-fragment K order, pack to bf16 pairs:
        // lane<16 needs K {0..7, 16..23}; lane>=16 needs K {8..15, 24..31}
        int base0 = half ? 8 : 0;
        const float* hr = h + (size_t)src * HH;
#pragma unroll
        for (int v = 0; v < 4; v++)
            hspk[v] = cvt_pk_bf16(hr[base0 + 2 * v], hr[base0 + 2 * v + 1]);
#pragma unroll
        for (int v = 0; v < 4; v++)
            hspk[4 + v] = cvt_pk_bf16(hr[16 + base0 + 2 * v], hr[16 + base0 + 2 * v + 1]);
    }

    if (wave == 0) {
        __builtin_amdgcn_s_wait_tensorcnt(0);   // LDS contents of sB now valid
    }
    __syncthreads();
    if (!active) return;

    union AU { v16bf v; uint32_t u[8]; };
    union BU { v16bf v; uint4 q[2]; };

    v8f acc0 = {};   // columns o = 0..15
    v8f acc1 = {};   // columns o = 16..31

    // ---- bias chunk: msg += hs @ b2  (A = hs, t == 1) ----
    {
        AU a;
#pragma unroll
        for (int v = 0; v < 8; v++) a.u[v] = hspk[v];
        const uint4* bb = (const uint4*)(BswBias + (size_t)layer * (2 * 32 * 8));
        BU b0, b1u;
        b0.q[0]  = bb[lane * 2 + 0];        b0.q[1]  = bb[lane * 2 + 1];
        b1u.q[0] = bb[(32 + lane) * 2 + 0]; b1u.q[1] = bb[(32 + lane) * 2 + 1];
        acc0 = __builtin_amdgcn_wmma_f32_16x16x32_bf16(false, a.v, false, b0.v,  (short)0, acc0, false, false);
        acc1 = __builtin_amdgcn_wmma_f32_16x16x32_bf16(false, a.v, false, b1u.v, (short)0, acc1, false, false);
    }

    // ---- main contraction: 32 K-chunks (chunk kk == fixed kt), A = t[kk]*hs ----
    // A built with 8 v_pk_mul_bf16 per chunk; B from LDS (ds_load_b128 x4).
    const uint4* s4 = (const uint4*)sB;
#pragma unroll
    for (int kk = 0; kk < 32; kk++) {
        AU a;
#pragma unroll
        for (int v = 0; v < 8; v++) a.u[v] = pk_mul_bf16(hspk[v], tspk[kk]);
        int base_nt0 = (kk * 2 + 0) * 64 + lane * 2;
        int base_nt1 = (kk * 2 + 1) * 64 + lane * 2;
        BU b0, b1u;
        b0.q[0]  = s4[base_nt0];     b0.q[1]  = s4[base_nt0 + 1];
        b1u.q[0] = s4[base_nt1];     b1u.q[1] = s4[base_nt1 + 1];
        acc0 = __builtin_amdgcn_wmma_f32_16x16x32_bf16(false, a.v, false, b0.v,  (short)0, acc0, false, false);
        acc1 = __builtin_amdgcn_wmma_f32_16x16x32_bf16(false, a.v, false, b1u.v, (short)0, acc1, false, false);
    }

    // ---- scatter-add: D layout lane L, VGPR r -> edge = r + 8*(L>=16), col = L%16 ----
    int hi8 = half ? 8 : 0;
    int col = lane & 15;
#pragma unroll
    for (int r = 0; r < 8; r++) {
        int eo = tile * 16 + r + hi8;
        int di = eo + EE;                 // dst = EIflat[(e+E) mod 2E]
        if (di >= E2) di -= E2;
        int d = EI[di];
        atomicAdd(&agg[(size_t)d * HH + col],      acc0[r]);
        atomicAdd(&agg[(size_t)d * HH + 16 + col], acc1[r]);
    }
}

// ---------------- node update: hc = agg + h@Wr + b; LayerNorm; h += gelu ----
__global__ void node_update_kernel(float* __restrict__ h,
                                   const float* __restrict__ agg,
                                   const float* __restrict__ W_root,
                                   const float* __restrict__ conv_b,
                                   const float* __restrict__ gamma,
                                   const float* __restrict__ beta,
                                   int layer) {
    int gid = blockIdx.x * blockDim.x + threadIdx.x;
    int n = gid >> 5, o = gid & 31;
    if (n >= NN) return;
    float hv = h[(size_t)n * HH + o];
    const float* wr = W_root + (size_t)layer * HH * HH;
    float acc = agg[(size_t)n * HH + o] + conv_b[layer * HH + o];
#pragma unroll
    for (int j = 0; j < HH; j++) acc += __shfl(hv, j, 32) * wr[j * HH + o];
    // LayerNorm over the 32 lanes
    float s = acc;
#pragma unroll
    for (int m = 16; m > 0; m >>= 1) s += __shfl_xor(s, m, 32);
    float mu = s * (1.0f / 32.0f);
    float d = acc - mu;
    float vs = d * d;
#pragma unroll
    for (int m = 16; m > 0; m >>= 1) vs += __shfl_xor(vs, m, 32);
    float var = vs * (1.0f / 32.0f);
    float hn = d * rsqrtf(var + 1e-5f) * gamma[layer * HH + o] + beta[layer * HH + o];
    h[(size_t)n * HH + o] = hv + gelu_exact(hn);
}

// ---------------- decoder: out = gelu(h@dW1+db1)@dW2 + db2 ----
__global__ void decode_kernel(const float* __restrict__ h,
                              const float* __restrict__ dW1,
                              const float* __restrict__ db1,
                              const float* __restrict__ dW2,
                              const float* __restrict__ db2,
                              float* __restrict__ out) {
    int gid = blockIdx.x * blockDim.x + threadIdx.x;
    int n = gid >> 5, o = gid & 31;
    if (n >= NN) return;
    float hv = h[(size_t)n * HH + o];
    float z = db1[o];
#pragma unroll
    for (int j = 0; j < HH; j++) z += __shfl(hv, j, 32) * dW1[j * HH + o];
    float g = gelu_exact(z);
#pragma unroll
    for (int fo = 0; fo < FOO; fo++) {
        float p = g * dW2[o * FOO + fo];
#pragma unroll
        for (int m = 16; m > 0; m >>= 1) p += __shfl_xor(p, m, 32);
        if (o == 0) out[(size_t)n * FOO + fo] = p + db2[fo];
    }
}

// ---------------- launch ----------------
extern "C" void kernel_launch(void* const* d_in, const int* in_sizes, int n_in,
                              void* d_out, int out_size, void* d_ws, size_t ws_size,
                              hipStream_t stream) {
    const float* x       = (const float*)d_in[0];
    const float* pm      = (const float*)d_in[1];
    const int*   EI      = (const int*)d_in[2];      // [2,E] int32
    const float* ea      = (const float*)d_in[3];
    const float* W_in    = (const float*)d_in[4];
    const float* b_in    = (const float*)d_in[5];
    const float* mW1     = (const float*)d_in[6];
    const float* mb1     = (const float*)d_in[7];
    const float* mW2     = (const float*)d_in[8];
    const float* mb2     = (const float*)d_in[9];
    const float* eW1     = (const float*)d_in[10];
    const float* eb1     = (const float*)d_in[11];
    const float* eW2     = (const float*)d_in[12];
    const float* eb2     = (const float*)d_in[13];
    const float* W_root  = (const float*)d_in[14];
    const float* conv_b  = (const float*)d_in[15];
    const float* gamma   = (const float*)d_in[16];
    const float* beta    = (const float*)d_in[17];
    const float* dW1     = (const float*)d_in[18];
    const float* db1     = (const float*)d_in[19];
    const float* dW2     = (const float*)d_in[20];
    const float* db2     = (const float*)d_in[21];
    float* out = (float*)d_out;

    // workspace layout
    float*    h    = (float*)d_ws;
    float*    agg  = h + (size_t)NN * HH;
    uint32_t* Bsw  = (uint32_t*)(agg + (size_t)NN * HH);
    uint32_t* Bbia = Bsw + BSW_MAIN_DW;

    // 1) pre-swizzle w2/b2 into bf16 B-operand fragments
    {
        int tot = BSW_MAIN_DW + BSW_BIAS_DW;
        prep_b_kernel<<<(tot + 255) / 256, 256, 0, stream>>>(eW2, eb2, Bsw, Bbia);
    }
    // 2) input encoder
    node_init_kernel<<<(NN * HH) / 256, 256, 0, stream>>>(x, pm, W_in, b_in, mW1, mb1, mW2, mb2, h);

    // 3) three message-passing layers
    for (int l = 0; l < LL; l++) {
        zero_kernel<<<(NN * HH + 255) / 256, 256, 0, stream>>>(agg, NN * HH);
        edge_kernel<<<(TILES + 7) / 8, 256, 0, stream>>>(h, agg, EI, ea, eW1, eb1, Bsw, Bbia, l);
        node_update_kernel<<<(NN * HH) / 256, 256, 0, stream>>>(h, agg, W_root, conv_b, gamma, beta, l);
    }
    // 4) decoder
    decode_kernel<<<(NN * HH) / 256, 256, 0, stream>>>(h, dW1, db1, dW2, db2, out);
}
